// SimpleGAT_87780541595690
// MI455X (gfx1250) — compile-verified, matrix operand order checked by reading, not run on hardware
//
#include <hip/hip_runtime.h>
#include <hip/hip_bf16.h>

typedef _Float16 v16h __attribute__((ext_vector_type(16)));
typedef _Float16 h8   __attribute__((ext_vector_type(8)));
typedef float    v8f  __attribute__((ext_vector_type(8)));
typedef unsigned u32x4 __attribute__((ext_vector_type(4)));
typedef unsigned u32x8 __attribute__((ext_vector_type(8)));
typedef __attribute__((address_space(3))) _Float16 lds_f16;

#define NEG_SLOPE 0.2f

__device__ __forceinline__ unsigned fenc(float f) {
  unsigned u = __float_as_uint(f);
  return (u & 0x80000000u) ? ~u : (u | 0x80000000u);
}
__device__ __forceinline__ float fdec(unsigned e) {
  return (e & 0x80000000u) ? __uint_as_float(e ^ 0x80000000u) : __uint_as_float(~e);
}
__device__ __forceinline__ float lrelu(float x) { return x > 0.f ? x : NEG_SLOPE * x; }
__device__ __forceinline__ float elu1(float x)  { return x > 0.f ? x : (expf(x) - 1.f); }

// ---------------------------------------------------------------- zero fill
__global__ void k_zero(float* p, long n) {
  long i = (long)blockIdx.x * blockDim.x + threadIdx.x;
  if (i < n) p[i] = 0.f;
}

// ------------------------------------------- weight transpose+convert to f16
// W is [K x Nc] row-major f32; Bt becomes [Nc x K] row-major f16.
__global__ void k_w_to_h_t(const float* __restrict__ W, _Float16* __restrict__ Bt,
                           int K, int Nc) {
  int i = blockIdx.x * blockDim.x + threadIdx.x;
  if (i >= K * Nc) return;
  int n = i / K, k = i % K;
  Bt[i] = (_Float16)W[(size_t)k * Nc + n];
}

// ---------------------------------------------------------------- WMMA GEMM
// C[M,Nc] = A[M,K] @ B[K,Nc]; A f32, B pre-transposed f16 (Bt[Nc][K]), C f32.
// Block = 8 waves sharing one 16-col tile. The B panel (16 x K f16 = 8KB) is
// DMA'd into LDS by the Tensor Data Mover (tensor_load_to_lds, TENSORcnt),
// then each wave computes a 16x16 C tile with K fully unrolled
// (K/32 v_wmma_f32_16x16x32_f16 per wave).
// If xidx != nullptr, row r of A is embT + xidx[r]*K (embedding gather).
template <int K>
__global__ void k_gemm_wmma(const float* __restrict__ A, const _Float16* __restrict__ Bt,
                            float* __restrict__ C, int M, int Nc,
                            const int* __restrict__ xidx, const float* __restrict__ embT) {
  __shared__ _Float16 Bs[16 * K];

  const int tilesN = Nc >> 4;
  const int tilesM = M >> 4;
  const int bcol   = blockIdx.x % tilesN;
  const int rb     = blockIdx.x / tilesN;
  const int w      = threadIdx.x >> 5;
  const int lane   = threadIdx.x & 31;

  // ---- TDM: stage B panel Bt[bcol*16 .. +16)[0..K) f16 into LDS.
  // One wave issues the descriptor; TDM ignores EXEC; tracked by TENSORcnt.
  if (w == 0) {
    const unsigned long long ga =
        (unsigned long long)(Bt + (size_t)bcol * 16 * K);       // tile global addr
    const unsigned la = (unsigned)(unsigned long long)(lds_f16*)Bs;  // LDS byte addr

    u32x4 g0;
    g0[0] = 1u;                                   // count=1, user mode, no gather
    g0[1] = la;                                   // lds_addr
    g0[2] = (unsigned)(ga & 0xFFFFFFFFull);       // global_addr[31:0]
    g0[3] = (unsigned)((ga >> 32) & 0x01FFFFFFull) | (2u << 30);  // [56:32] | type=2

    u32x8 g1;
    g1[0] = 1u << 16;                             // wg_mask=0, data_size=1 (2B), no flags
    g1[1] = ((unsigned)K & 0xFFFFu) << 16;        // ab_addr=0 | tensor_dim0[15:0]
    g1[2] = (((unsigned)K >> 16) & 0xFFFFu) | (16u << 16);  // td0[31:16] | tensor_dim1[15:0]=16
    g1[3] = ((unsigned)K << 16);                  // td1[31:16]=0 | tile_dim0=K
    g1[4] = 16u;                                  // tile_dim1=16, tile_dim2=0
    g1[5] = (unsigned)K;                          // tensor_dim0_stride[31:0]
    g1[6] = 0u;                                   // stride0[47:32]=0, stride1 lo=0
    g1[7] = 0u;                                   // stride1 hi=0

    asm volatile("tensor_load_to_lds %0, %1" :: "s"(g0), "s"(g1) : "memory");
#if __has_builtin(__builtin_amdgcn_s_wait_tensorcnt)
    __builtin_amdgcn_s_wait_tensorcnt(0);
#else
    asm volatile("s_wait_tensorcnt 0x0" ::: "memory");
#endif
  }
  __syncthreads();

  const int tmTile = rb * 8 + w;
  if (tmTile >= tilesM) return;                 // wave-uniform; EXEC stays all-ones
  const int tm = tmTile << 4;
  const int tn = bcol << 4;

  const int mr = lane & 15;                     // A row in tile / B,C col in tile
  const int kb = (lane >> 4) << 3;              // 0 or 8: K sub-block per half-wave
  const int r  = tm + mr;
  const float* arow = xidx ? (embT + (size_t)xidx[r] * K) : (A + (size_t)r * K);
  const _Float16* brow = Bs + mr * K;

  v8f acc = {};
#pragma unroll
  for (int k0 = 0; k0 < K; k0 += 32) {
    // A fragment: lanes 0-15 hold K=k0+0..7,16..23; lanes 16-31 hold 8..15,24..31
    const float4 a0 = *(const float4*)(arow + k0 + kb);
    const float4 a1 = *(const float4*)(arow + k0 + kb + 4);
    const float4 a2 = *(const float4*)(arow + k0 + kb + 16);
    const float4 a3 = *(const float4*)(arow + k0 + kb + 20);
    v16h af;
    af[0]  = (_Float16)a0.x; af[1]  = (_Float16)a0.y; af[2]  = (_Float16)a0.z; af[3]  = (_Float16)a0.w;
    af[4]  = (_Float16)a1.x; af[5]  = (_Float16)a1.y; af[6]  = (_Float16)a1.z; af[7]  = (_Float16)a1.w;
    af[8]  = (_Float16)a2.x; af[9]  = (_Float16)a2.y; af[10] = (_Float16)a2.z; af[11] = (_Float16)a2.w;
    af[12] = (_Float16)a3.x; af[13] = (_Float16)a3.y; af[14] = (_Float16)a3.z; af[15] = (_Float16)a3.w;
    // B fragment: same K pattern, column tn+mr, contiguous f16 from LDS
    const h8 b0 = *(const h8*)(brow + k0 + kb);
    const h8 b1 = *(const h8*)(brow + k0 + kb + 16);
    v16h bf;
#pragma unroll
    for (int i = 0; i < 8; ++i) { bf[i] = b0[i]; bf[i + 8] = b1[i]; }
    acc = __builtin_amdgcn_wmma_f32_16x16x32_f16(false, af, false, bf,
                                                 (short)0, acc, false, false);
  }
  // C/D layout: VGPR v -> row v (lanes 0-15) / row v+8 (lanes 16-31), col = lane&15
  const int mb = (lane >> 4) << 3;
  float* crow = C + (size_t)(tm + mb) * Nc + tn + mr;
#pragma unroll
  for (int v = 0; v < 8; ++v) crow[(size_t)v * Nc] = acc[v];
}

// ------------------------------------------------- attention coefficients
__global__ void k_attn_coef(const float* __restrict__ H, const float* __restrict__ as,
                            const float* __restrict__ ad, float* AS, float* AD,
                            int n, int heads, int hid) {
  int i = blockIdx.x * blockDim.x + threadIdx.x;
  if (i >= n) return;
  const float* row = H + (size_t)i * heads * hid;
  for (int h = 0; h < heads; ++h) {
    float s = 0.f, d = 0.f;
    for (int k = 0; k < hid; ++k) {
      float v = row[h * hid + k];
      s += v * as[h * hid + k];
      d += v * ad[h * hid + k];
    }
    AS[i * heads + h] = s;
    AD[i * heads + h] = d;
  }
}

// ------------------------------------------------- segment softmax: max pass
__global__ void k_edge_max(const int* __restrict__ ei, int E, int N,
                           const float* __restrict__ AS, const float* __restrict__ AD,
                           unsigned* M, int heads) {
  int e = blockIdx.x * blockDim.x + threadIdx.x;
  int Etot = E + N;
  if (e >= Etot) return;
  int s = (e < E) ? ei[e]     : (e - E);
  int d = (e < E) ? ei[E + e] : (e - E);
  for (int h = 0; h < heads; ++h) {
    float v = lrelu(AS[s * heads + h] + AD[d * heads + h]);
    atomicMax(&M[d * heads + h], fenc(v));
  }
}

// ------------------------------------------------- segment softmax: denom pass
__global__ void k_edge_den(const int* __restrict__ ei, int E, int N,
                           const float* __restrict__ AS, const float* __restrict__ AD,
                           const unsigned* __restrict__ M, float* DEN, int heads) {
  int e = blockIdx.x * blockDim.x + threadIdx.x;
  int Etot = E + N;
  if (e >= Etot) return;
  int s = (e < E) ? ei[e]     : (e - E);
  int d = (e < E) ? ei[E + e] : (e - E);
  for (int h = 0; h < heads; ++h) {
    float v = lrelu(AS[s * heads + h] + AD[d * heads + h]);
    atomicAdd(&DEN[d * heads + h], expf(v - fdec(M[d * heads + h])));
  }
}

// ------------------------------------------------- weighted feature scatter
// One wave per edge; 32 lanes stripe the heads*hid features (coalesced).
__global__ void k_edge_scatter(const int* __restrict__ ei, int E, int N,
                               const float* __restrict__ AS, const float* __restrict__ AD,
                               const unsigned* __restrict__ M, const float* __restrict__ DEN,
                               const float* __restrict__ H, float* __restrict__ OUT,
                               int heads, int hid) {
  int w    = (blockIdx.x * blockDim.x + threadIdx.x) >> 5;
  int lane = threadIdx.x & 31;
  int Etot = E + N;
  if (w >= Etot) return;
  int s = (w < E) ? ei[w]     : (w - E);
  int d = (w < E) ? ei[E + w] : (w - E);
  float alpha[4];
  for (int h = 0; h < heads; ++h) {
    float v = lrelu(AS[s * heads + h] + AD[d * heads + h]);
    alpha[h] = expf(v - fdec(M[d * heads + h])) / (DEN[d * heads + h] + 1e-16f);
  }
  int F = heads * hid;
  const float* hs = H + (size_t)s * F;
  float*       od = OUT + (size_t)d * F;
  for (int f = lane; f < F; f += 32)
    atomicAdd(&od[f], hs[f] * alpha[f / hid]);
}

// ------------------------------------------------- bias + ELU (in place)
__global__ void k_bias_elu(float* X, const float* __restrict__ b, long n, int F) {
  long i = (long)blockIdx.x * blockDim.x + threadIdx.x;
  if (i >= n) return;
  X[i] = elu1(X[i] + b[i % F]);
}

// ------------------------------------------------- global mean pool (sums/cnt)
__global__ void k_pool(const float* __restrict__ X, const int* __restrict__ batch,
                       float* SUM, float* CNT, int n, int hid) {
  long i = (long)blockIdx.x * blockDim.x + threadIdx.x;
  int node = (int)(i / hid), f = (int)(i % hid);
  if (node >= n) return;
  int g = batch[node];
  atomicAdd(&SUM[(size_t)g * hid + f], X[(size_t)node * hid + f]);
  if (f == 0) atomicAdd(&CNT[g], 1.f);
}

// ------------------------------------------------- final FC: out = pooled@W + b
__global__ void k_final(const float* __restrict__ SUM, const float* __restrict__ CNT,
                        const float* __restrict__ W, const float* __restrict__ b,
                        float* __restrict__ out, int G, int hid, int od) {
  int i = blockIdx.x * blockDim.x + threadIdx.x;
  if (i >= G * od) return;
  int g = i / od, o = i % od;
  float c = fmaxf(CNT[g], 1.f);
  float acc = 0.f;
  for (int k = 0; k < hid; ++k) acc += (SUM[g * hid + k] / c) * W[k * od + o];
  out[i] = acc + b[o];
}

extern "C" void kernel_launch(void* const* d_in, const int* in_sizes, int n_in,
                              void* d_out, int out_size, void* d_ws, size_t ws_size,
                              hipStream_t stream) {
  const int*   x     = (const int*)  d_in[0];
  const int*   ei    = (const int*)  d_in[1];
  const int*   batch = (const int*)  d_in[2];
  const float* emb   = (const float*)d_in[3];
  const float* W1    = (const float*)d_in[4];
  const float* a1s   = (const float*)d_in[5];
  const float* a1d   = (const float*)d_in[6];
  const float* b1    = (const float*)d_in[7];
  const float* W2    = (const float*)d_in[8];
  const float* a2s   = (const float*)d_in[9];
  const float* a2d   = (const float*)d_in[10];
  const float* b2    = (const float*)d_in[11];
  const float* fcw   = (const float*)d_in[12];
  const float* fcb   = (const float*)d_in[13];
  float* out = (float*)d_out;

  const int N = in_sizes[0];
  const int E = in_sizes[1] / 2;
  const int IN = 256, HID = 64, HEADS = 4, F1 = HEADS * HID, ODIM = 64;
  const int G = out_size / ODIM;
  const int Etot = E + N;
  (void)n_in; (void)ws_size;

  float* ws = (float*)d_ws;
  size_t off = 0;
  auto alloc = [&](size_t cnt) { size_t o = off; off += (cnt + 63) & ~(size_t)63; return o; };
  // zero-initialized region (accumulators + max/den state), contiguous:
  size_t oX1  = alloc((size_t)N * F1);
  size_t oM1  = alloc((size_t)N * HEADS);
  size_t oD1  = alloc((size_t)N * HEADS);
  size_t oX2  = alloc((size_t)N * HID);
  size_t oM2  = alloc((size_t)N);
  size_t oD2  = alloc((size_t)N);
  size_t oSUM = alloc((size_t)G * HID);
  size_t oCNT = alloc((size_t)G);
  long   zcnt = (long)off;
  // uninitialized scratch:
  size_t oH1  = alloc((size_t)N * F1);
  size_t oAS1 = alloc((size_t)N * HEADS);
  size_t oAD1 = alloc((size_t)N * HEADS);
  size_t oH2  = alloc((size_t)N * HID);
  size_t oAS2 = alloc((size_t)N);
  size_t oAD2 = alloc((size_t)N);
  size_t oW1h = alloc((size_t)IN * F1 / 2 + 64);   // f16 [F1][IN]
  size_t oW2h = alloc((size_t)F1 * HID / 2 + 64);  // f16 [HID][F1]
  _Float16* W1h = (_Float16*)(ws + oW1h);
  _Float16* W2h = (_Float16*)(ws + oW2h);

  dim3 blk(256);
  auto cdiv = [](long a, long b) { return (unsigned)((a + b - 1) / b); };

  k_zero<<<cdiv(zcnt, 256), blk, 0, stream>>>(ws, zcnt);
  k_w_to_h_t<<<cdiv((long)IN * F1, 256), blk, 0, stream>>>(W1, W1h, IN, F1);
  k_w_to_h_t<<<cdiv((long)F1 * HID, 256), blk, 0, stream>>>(W2, W2h, F1, HID);

  // ---- layer 1: H1 = emb[x] @ W1  (embedding gather fused into GEMM)
  {
    int tilesN = F1 / 16;
    int rowBlocks = (N / 16 + 7) / 8;
    k_gemm_wmma<256><<<rowBlocks * tilesN, blk, 0, stream>>>(
        nullptr, W1h, ws + oH1, N, F1, x, emb);
  }
  k_attn_coef<<<cdiv(N, 256), blk, 0, stream>>>(ws + oH1, a1s, a1d,
                                                ws + oAS1, ws + oAD1, N, HEADS, HID);
  k_edge_max<<<cdiv(Etot, 256), blk, 0, stream>>>(ei, E, N, ws + oAS1, ws + oAD1,
                                                  (unsigned*)(ws + oM1), HEADS);
  k_edge_den<<<cdiv(Etot, 256), blk, 0, stream>>>(ei, E, N, ws + oAS1, ws + oAD1,
                                                  (unsigned*)(ws + oM1), ws + oD1, HEADS);
  k_edge_scatter<<<cdiv((long)Etot * 32, 256), blk, 0, stream>>>(
      ei, E, N, ws + oAS1, ws + oAD1, (unsigned*)(ws + oM1), ws + oD1,
      ws + oH1, ws + oX1, HEADS, HID);
  k_bias_elu<<<cdiv((long)N * F1, 256), blk, 0, stream>>>(ws + oX1, b1, (long)N * F1, F1);

  // ---- layer 2: H2 = X1 @ W2 (heads=1; mean over 1 head == identity)
  {
    int tilesN = HID / 16;
    int rowBlocks = (N / 16 + 7) / 8;
    k_gemm_wmma<256><<<rowBlocks * tilesN, blk, 0, stream>>>(
        ws + oX1, W2h, ws + oH2, N, HID, nullptr, nullptr);
  }
  k_attn_coef<<<cdiv(N, 256), blk, 0, stream>>>(ws + oH2, a2s, a2d,
                                                ws + oAS2, ws + oAD2, N, 1, HID);
  k_edge_max<<<cdiv(Etot, 256), blk, 0, stream>>>(ei, E, N, ws + oAS2, ws + oAD2,
                                                  (unsigned*)(ws + oM2), 1);
  k_edge_den<<<cdiv(Etot, 256), blk, 0, stream>>>(ei, E, N, ws + oAS2, ws + oAD2,
                                                  (unsigned*)(ws + oM2), ws + oD2, 1);
  k_edge_scatter<<<cdiv((long)Etot * 32, 256), blk, 0, stream>>>(
      ei, E, N, ws + oAS2, ws + oAD2, (unsigned*)(ws + oM2), ws + oD2,
      ws + oH2, ws + oX2, 1, HID);
  k_bias_elu<<<cdiv((long)N * HID, 256), blk, 0, stream>>>(ws + oX2, b2, (long)N * HID, HID);

  // ---- global mean pool + final FC
  k_pool<<<cdiv((long)N * HID, 256), blk, 0, stream>>>(ws + oX2, batch,
                                                       ws + oSUM, ws + oCNT, N, HID);
  k_final<<<cdiv((long)G * ODIM, 256), blk, 0, stream>>>(ws + oSUM, ws + oCNT,
                                                         fcw, fcb, out, G, HID, ODIM);
}
